// CustomGRUModel_89215060673052
// MI455X (gfx1250) — matrix operand order repaired and checked by reading, not compiled, not run on hardware
//
#include <hip/hip_runtime.h>

// ---------------------------------------------------------------------------
// Complex 2-layer GRU for MI455X (gfx1250), bf16 WMMA + fp32 accumulate.
// B=128, T=128, IN=256, H=2048, OUT=64; m=H/2=1024, n0=1152, n1=2048.
// Each wave computes a 32-row x 16-col (real,imag) output tile pair:
// weight (B) fragments reused across two batch sub-tiles -> 8 WMMA / 12
// 16B loads per k-step.  Blocks pack 2 batch-pairs x 4 m-tiles so both A and
// B fragments get WGP$ reuse and weight L2 duplication drops to 2x.
// ---------------------------------------------------------------------------

typedef __attribute__((ext_vector_type(16))) __bf16 v16bf;
typedef __attribute__((ext_vector_type(8)))  float  v8f;

#define B_   128
#define T_   128
#define IN_  256
#define H_   2048
#define M_   1024     // H/2 (complex half)
#define N0_  1152     // (IN+H)/2
#define N1_  2048     // (H+H)/2

static __device__ __forceinline__ unsigned short f32_to_bf16(float f) {
  unsigned int u = __float_as_uint(f);
  u += 0x7FFFu + ((u >> 16) & 1u);          // round-to-nearest-even
  return (unsigned short)(u >> 16);
}

// Build a 16-element bf16 fragment from two 16-byte chunks.
static __device__ __forceinline__ v16bf ld_frag(const unsigned short* p1,
                                                const unsigned short* p2) {
  union { uint4 u[2]; v16bf v; } c;
  c.u[0] = *(const uint4*)p1;
  c.u[1] = *(const uint4*)p2;
  return c.v;
}

static __device__ __forceinline__ v8f wmma_bf16(v16bf a, v16bf b, v8f c) {
  // 8 args: (neg_a, A, neg_b, B, c_mod, C, reuse_a, reuse_b)
  return __builtin_amdgcn_wmma_f32_16x16x32_bf16(false, a, false, b,
                                                 (short)0, c, false, false);
}

// Accumulator bundle for one 32x16 complex-linear tile pair.
struct Acc6 {
  v8f rp0, rn0, i0;   // batch rows [b0 .. b0+15]
  v8f rp1, rn1, i1;   // batch rows [b0+16 .. b0+31]
};

static __device__ __forceinline__ void kstep(
    Acc6& a, const unsigned short* ar0, const unsigned short* ai0,
    const unsigned short* ar1, const unsigned short* ai1,
    const unsigned short* wr, const unsigned short* wi, int ksub) {
  v16bf b_wr = ld_frag(wr, wr + 8);
  v16bf b_wi = ld_frag(wi, wi + 8);
  v16bf a_cr0 = ld_frag(ar0 + 2 * ksub, ar0 + 8 + 2 * ksub);
  v16bf a_ci0 = ld_frag(ai0 + 2 * ksub, ai0 + 8 + 2 * ksub);
  v16bf a_cr1 = ld_frag(ar1 + 2 * ksub, ar1 + 8 + 2 * ksub);
  v16bf a_ci1 = ld_frag(ai1 + 2 * ksub, ai1 + 8 + 2 * ksub);
  a.rp0 = wmma_bf16(a_cr0, b_wr, a.rp0);
  a.rn0 = wmma_bf16(a_ci0, b_wi, a.rn0);
  a.i0  = wmma_bf16(a_cr0, b_wi, a.i0);
  a.i0  = wmma_bf16(a_ci0, b_wr, a.i0);
  a.rp1 = wmma_bf16(a_cr1, b_wr, a.rp1);
  a.rn1 = wmma_bf16(a_ci1, b_wi, a.rn1);
  a.i1  = wmma_bf16(a_cr1, b_wi, a.i1);
  a.i1  = wmma_bf16(a_ci1, b_wr, a.i1);
}

// ---------------------------------------------------------------------------
// fp32 -> bf16 conversion, 4 elements/thread (float4 load, uint2 store)
// ---------------------------------------------------------------------------
__global__ void cvt_bf16_kernel(const float* __restrict__ src,
                                unsigned short* __restrict__ dst, int n4) {
  int i = blockIdx.x * blockDim.x + threadIdx.x;
  int stride = gridDim.x * blockDim.x;
  const float4* s4 = (const float4*)src;
  uint2* d2 = (uint2*)dst;
  for (; i < n4; i += stride) {
    float4 v = s4[i];
    uint2 o;
    o.x = (unsigned int)f32_to_bf16(v.x) | ((unsigned int)f32_to_bf16(v.y) << 16);
    o.y = (unsigned int)f32_to_bf16(v.z) | ((unsigned int)f32_to_bf16(v.w) << 16);
    d2[i] = o;
  }
}

// zero h0/h1 (fp32 + bf16 shadows)
__global__ void zero_h_kernel(float* __restrict__ h0, float* __restrict__ h1,
                              unsigned short* __restrict__ h0b,
                              unsigned short* __restrict__ h1b, int n) {
  int i = blockIdx.x * blockDim.x + threadIdx.x;
  int stride = gridDim.x * blockDim.x;
  for (; i < n; i += stride) {
    h0[i] = 0.f; h1[i] = 0.f; h0b[i] = 0; h1b[i] = 0;
  }
}

// ---------------------------------------------------------------------------
// Gates kernel: computes gates 0 (z) and 1 (r), writes z (fp32, Bx2048) and
// tmp = r*h (bf16, Bx2048).
//   zr = cr@Wr.T - ci@Wi.T + br   (accRp - accRn)
//   zi = cr@Wi.T + ci@Wr.T + bi   (accI)
// 64 blocks = gate(2) x batch-quad(2) x m-group(16); 8 waves/block =
// 2 batch-pairs x 4 m-tiles (A and B fragments WGP$-shared).
// ---------------------------------------------------------------------------
__global__ __launch_bounds__(256) void gates_kernel(
    const unsigned short* __restrict__ xact, int xRowStride, int xColOff, int ih,
    const unsigned short* __restrict__ hbf,
    const unsigned short* __restrict__ Wr,
    const unsigned short* __restrict__ Wi, int n,
    const float* __restrict__ br, const float* __restrict__ bi,
    const float* __restrict__ h,
    float* __restrict__ z_out, unsigned short* __restrict__ tmp_out) {
  const int wave = threadIdx.x >> 5;
  const int lane = threadIdx.x & 31;
  const int blk  = blockIdx.x;                       // 0..63
  const int g    = blk >> 5;                         // gate 0 or 1
  const int b0   = (((blk >> 4) & 1) * 2 + (wave >> 2)) * 32;  // 32-row base
  const int m0   = ((blk & 15) * 4 + (wave & 3)) * 16;         // col tile

  const int ln15 = lane & 15;
  const int hi16 = lane & 16;                  // 0 or 16
  const int ksub = hi16 >> 1;                  // 0 or 8

  const unsigned short* xrow0 =
      xact + (size_t)(b0 + ln15) * xRowStride + xColOff;
  const unsigned short* xrow1 =
      xact + (size_t)(b0 + 16 + ln15) * xRowStride + xColOff;
  const unsigned short* hrow0 = hbf + (size_t)(b0 + ln15) * H_;
  const unsigned short* hrow1 = hbf + (size_t)(b0 + 16 + ln15) * H_;

  const unsigned short* wr_row =
      Wr + (size_t)(g * M_ + m0 + ln15) * n + hi16;   // B col = W row (m)
  const unsigned short* wi_row =
      Wi + (size_t)(g * M_ + m0 + ln15) * n + hi16;

  Acc6 acc = {};

  // x-part of cr/ci  (k in [0, ih))
  for (int k0 = 0; k0 < ih; k0 += 32) {
    kstep(acc, xrow0 + k0, xrow0 + ih + k0, xrow1 + k0, xrow1 + ih + k0,
          wr_row + k0, wi_row + k0, ksub);
  }
  // h-part of cr/ci  (k in [ih, n))
  const unsigned short* wr_h = wr_row + ih;
  const unsigned short* wi_h = wi_row + ih;
  const int nh = n - ih;
  for (int k0 = 0; k0 < nh; k0 += 32) {
    kstep(acc, hrow0 + k0, hrow0 + M_ + k0, hrow1 + k0, hrow1 + M_ + k0,
          wr_h + k0, wi_h + k0, ksub);
  }

  const int m  = m0 + ln15;
  const float brv = br[g * M_ + m];
  const float biv = bi[g * M_ + m];

#pragma unroll
  for (int st = 0; st < 2; ++st) {
    const v8f& rp = st ? acc.rp1 : acc.rp0;
    const v8f& rn = st ? acc.rn1 : acc.rn0;
    const v8f& ii = st ? acc.i1  : acc.i0;
#pragma unroll
    for (int j = 0; j < 8; ++j) {
      const int b = b0 + st * 16 + j + (hi16 ? 8 : 0);
      const float re = rp[j] - rn[j] + brv;
      const float im = ii[j] + biv;
      const float sr = 1.f / (1.f + __expf(-re));
      const float si = 1.f / (1.f + __expf(-im));
      if (g == 0) {                       // z gate
        z_out[(size_t)b * H_ + m]      = sr;
        z_out[(size_t)b * H_ + M_ + m] = si;
      } else {                            // r gate -> tmp = r*h (bf16)
        const float tr = sr * h[(size_t)b * H_ + m];
        const float ti = si * h[(size_t)b * H_ + M_ + m];
        tmp_out[(size_t)b * H_ + m]      = f32_to_bf16(tr);
        tmp_out[(size_t)b * H_ + M_ + m] = f32_to_bf16(ti);
      }
    }
  }
}

// ---------------------------------------------------------------------------
// Update kernel: gate 2 (candidate) with A h-part = tmp (=r*h), then
//   h_new = (1-z)*h + z*tanh(.)   ; writes h fp32 and its bf16 shadow.
// 32 blocks = batch-quad(2) x m-group(16); 8 waves = 2 bp x 4 mt.
// ---------------------------------------------------------------------------
__global__ __launch_bounds__(256) void update_kernel(
    const unsigned short* __restrict__ xact, int xRowStride, int xColOff, int ih,
    const unsigned short* __restrict__ tmp_bf,
    const unsigned short* __restrict__ Wr,
    const unsigned short* __restrict__ Wi, int n,
    const float* __restrict__ br, const float* __restrict__ bi,
    const float* __restrict__ z,
    float* __restrict__ h, unsigned short* __restrict__ hbf) {
  const int wave = threadIdx.x >> 5;
  const int lane = threadIdx.x & 31;
  const int blk  = blockIdx.x;                       // 0..31
  const int b0   = (((blk >> 4) & 1) * 2 + (wave >> 2)) * 32;
  const int m0   = ((blk & 15) * 4 + (wave & 3)) * 16;

  const int ln15 = lane & 15;
  const int hi16 = lane & 16;
  const int ksub = hi16 >> 1;

  const unsigned short* xrow0 =
      xact + (size_t)(b0 + ln15) * xRowStride + xColOff;
  const unsigned short* xrow1 =
      xact + (size_t)(b0 + 16 + ln15) * xRowStride + xColOff;
  const unsigned short* trow0 = tmp_bf + (size_t)(b0 + ln15) * H_;
  const unsigned short* trow1 = tmp_bf + (size_t)(b0 + 16 + ln15) * H_;

  const unsigned short* wr_row =
      Wr + (size_t)(2 * M_ + m0 + ln15) * n + hi16;   // gate 2
  const unsigned short* wi_row =
      Wi + (size_t)(2 * M_ + m0 + ln15) * n + hi16;

  Acc6 acc = {};

  for (int k0 = 0; k0 < ih; k0 += 32) {
    kstep(acc, xrow0 + k0, xrow0 + ih + k0, xrow1 + k0, xrow1 + ih + k0,
          wr_row + k0, wi_row + k0, ksub);
  }
  const unsigned short* wr_h = wr_row + ih;
  const unsigned short* wi_h = wi_row + ih;
  const int nh = n - ih;
  for (int k0 = 0; k0 < nh; k0 += 32) {
    kstep(acc, trow0 + k0, trow0 + M_ + k0, trow1 + k0, trow1 + M_ + k0,
          wr_h + k0, wi_h + k0, ksub);
  }

  const int m  = m0 + ln15;
  const float brv = br[2 * M_ + m];
  const float biv = bi[2 * M_ + m];

#pragma unroll
  for (int st = 0; st < 2; ++st) {
    const v8f& rp = st ? acc.rp1 : acc.rp0;
    const v8f& rn = st ? acc.rn1 : acc.rn0;
    const v8f& ii = st ? acc.i1  : acc.i0;
#pragma unroll
    for (int j = 0; j < 8; ++j) {
      const int b = b0 + st * 16 + j + (hi16 ? 8 : 0);
      const size_t ir = (size_t)b * H_ + m;
      const size_t ic = (size_t)b * H_ + M_ + m;
      const float ntr = tanhf(rp[j] - rn[j] + brv);
      const float nti = tanhf(ii[j] + biv);
      const float zr = z[ir], zi = z[ic];
      const float nr = (1.f - zr) * h[ir] + zr * ntr;
      const float ni = (1.f - zi) * h[ic] + zi * nti;
      h[ir] = nr;  h[ic] = ni;
      hbf[ir] = f32_to_bf16(nr);
      hbf[ic] = f32_to_bf16(ni);
    }
  }
}

// ---------------------------------------------------------------------------
// Final FC (tiny: 128x64 out, K=1024): fp32 dot products, one thread per out.
// ---------------------------------------------------------------------------
__global__ __launch_bounds__(256) void fc_kernel(
    const float* __restrict__ h1,
    const float* __restrict__ fcWr, const float* __restrict__ fcWi,
    const float* __restrict__ fcbr, const float* __restrict__ fcbi,
    float* __restrict__ out) {
  const int idx = blockIdx.x * blockDim.x + threadIdx.x;   // 0..8191
  if (idx >= B_ * 64) return;
  const int b = idx >> 6;
  const int c = idx & 63;
  const int j = c & 31;
  const bool imag = (c >= 32);
  const float* hr = h1 + (size_t)b * H_;
  const float* hi = hr + M_;
  const float* wr = fcWr + (size_t)j * M_;
  const float* wi = fcWi + (size_t)j * M_;
  float acc = 0.f;
  if (!imag) {
    for (int mm = 0; mm < M_; ++mm)
      acc = fmaf(hr[mm], wr[mm], fmaf(-hi[mm], wi[mm], acc));
    acc += fcbr[j];
  } else {
    for (int mm = 0; mm < M_; ++mm)
      acc = fmaf(hr[mm], wi[mm], fmaf(hi[mm], wr[mm], acc));
    acc += fcbi[j];
  }
  out[(size_t)b * 64 + c] = acc;
}

// ---------------------------------------------------------------------------
// Host-side orchestration
// ---------------------------------------------------------------------------
extern "C" void kernel_launch(void* const* d_in, const int* in_sizes, int n_in,
                              void* d_out, int out_size, void* d_ws, size_t ws_size,
                              hipStream_t stream) {
  (void)in_sizes; (void)n_in; (void)out_size; (void)ws_size;

  const float* x    = (const float*)d_in[0];
  const float* W0r  = (const float*)d_in[1];
  const float* W0i  = (const float*)d_in[2];
  const float* b0r  = (const float*)d_in[3];
  const float* b0i  = (const float*)d_in[4];
  const float* W1r  = (const float*)d_in[5];
  const float* W1i  = (const float*)d_in[6];
  const float* b1r  = (const float*)d_in[7];
  const float* b1i  = (const float*)d_in[8];
  const float* fcWr = (const float*)d_in[9];
  const float* fcWi = (const float*)d_in[10];
  const float* fcbr = (const float*)d_in[11];
  const float* fcbi = (const float*)d_in[12];
  float* out = (float*)d_out;

  const size_t NW0 = (size_t)3 * M_ * N0_;   // 3,538,944
  const size_t NW1 = (size_t)3 * M_ * N1_;   // 6,291,456
  const size_t NX  = (size_t)B_ * T_ * IN_;  // 4,194,304
  const size_t NH  = (size_t)B_ * H_;        //   262,144

  char* ws = (char*)d_ws;
  unsigned short* w0r_bf = (unsigned short*)ws; ws += NW0 * 2;
  unsigned short* w0i_bf = (unsigned short*)ws; ws += NW0 * 2;
  unsigned short* w1r_bf = (unsigned short*)ws; ws += NW1 * 2;
  unsigned short* w1i_bf = (unsigned short*)ws; ws += NW1 * 2;
  unsigned short* x_bf   = (unsigned short*)ws; ws += NX  * 2;
  float*          h0_f   = (float*)ws;          ws += NH  * 4;
  float*          h1_f   = (float*)ws;          ws += NH  * 4;
  unsigned short* h0_bf  = (unsigned short*)ws; ws += NH  * 2;
  unsigned short* h1_bf  = (unsigned short*)ws; ws += NH  * 2;
  float*          z_buf  = (float*)ws;          ws += NH  * 4;
  unsigned short* tmp_bf = (unsigned short*)ws; ws += NH  * 2;

  // 1) one-time (per launch) conversions: weights + x to bf16; zero h state
  cvt_bf16_kernel<<<1024, 256, 0, stream>>>(W0r, w0r_bf, (int)(NW0 / 4));
  cvt_bf16_kernel<<<1024, 256, 0, stream>>>(W0i, w0i_bf, (int)(NW0 / 4));
  cvt_bf16_kernel<<<1024, 256, 0, stream>>>(W1r, w1r_bf, (int)(NW1 / 4));
  cvt_bf16_kernel<<<1024, 256, 0, stream>>>(W1i, w1i_bf, (int)(NW1 / 4));
  cvt_bf16_kernel<<<1024, 256, 0, stream>>>(x,   x_bf,   (int)(NX  / 4));
  zero_h_kernel<<<512, 256, 0, stream>>>(h0_f, h1_f, h0_bf, h1_bf, (int)NH);

  // 2) sequential recurrence: 2 kernels per cell, 2 cells per timestep
  for (int t = 0; t < T_; ++t) {
    // layer 0: x-part = x[:, t, :], ih = 128, n = 1152
    gates_kernel<<<64, 256, 0, stream>>>(
        x_bf, T_ * IN_, t * IN_, IN_ / 2, h0_bf,
        w0r_bf, w0i_bf, N0_, b0r, b0i, h0_f, z_buf, tmp_bf);
    update_kernel<<<32, 256, 0, stream>>>(
        x_bf, T_ * IN_, t * IN_, IN_ / 2, tmp_bf,
        w0r_bf, w0i_bf, N0_, b0r, b0i, z_buf, h0_f, h0_bf);
    // layer 1: x-part = h0, ih = 1024, n = 2048
    gates_kernel<<<64, 256, 0, stream>>>(
        h0_bf, H_, 0, M_, h1_bf,
        w1r_bf, w1i_bf, N1_, b1r, b1i, h1_f, z_buf, tmp_bf);
    update_kernel<<<32, 256, 0, stream>>>(
        h0_bf, H_, 0, M_, tmp_bf,
        w1r_bf, w1i_bf, N1_, b1r, b1i, z_buf, h1_f, h1_bf);
  }

  // 3) final complex FC on h1
  fc_kernel<<<32, 256, 0, stream>>>(h1_f, fcWr, fcWi, fcbr, fcbi, out);
}
